// ComputeLoss_8151847928299
// MI455X (gfx1250) — compile-verified, matrix operand order checked by reading, not run on hardware
//
#include <hip/hip_runtime.h>
#include <hip/hip_bf16.h>
#include <cstdint>

// ---------------- problem constants ----------------
#define NB   8
#define NA   21504          // 128*128 + 64*64 + 32*32
#define NG   60
#define NCLS 15
#define GP   16             // class dim padded to 16 for WMMA
#define A_TILES 1344        // NA/16
#define NBLK_BA 672         // NB*NA/256

typedef __attribute__((ext_vector_type(2)))  float    v2f;
typedef __attribute__((ext_vector_type(8)))  float    v8f;
typedef __attribute__((ext_vector_type(16))) _Float16 v16h;

// ---------------- helpers ----------------
__device__ __forceinline__ void anchor_geo(int a, int& lvl, int& hw, int& w, int& ii, float& st){
  if (a < 16384)      { lvl=0; w=128; hw=16384; ii=a;        st=8.f;  }
  else if (a < 20480) { lvl=1; w=64;  hw=4096;  ii=a-16384;  st=16.f; }
  else                { lvl=2; w=32;  hw=1024;  ii=a-20480;  st=32.f; }
}

__device__ __forceinline__ void cov3(float w, float h, float t, float& a, float& b, float& c){
  float ct=cosf(t), s=sinf(t);
  float w2=w*w*(1.f/12.f), h2=h*h*(1.f/12.f);
  a = w2*ct*ct + h2*s*s;
  b = w2*s*s   + h2*ct*ct;
  c = (w2-h2)*ct*s;
}

// prob_iou with both covariances (and clamped dets) precomputed
__device__ __forceinline__ float prob_iou_pre(
    float x1,float y1,float a1,float b1,float c1,float d1,
    float x2,float y2,float a2,float b2,float c2,float d2){
  float dx=x1-x2, dy=y1-y2;
  float Av=a1+a2, Bv=b1+b2, Cv=c1+c2;
  float den = Av*Bv - Cv*Cv + 1e-8f;
  float t1 = 0.25f*(Av*dy*dy + Bv*dx*dx)/den;
  float t2 = 0.5f*(Cv*(-dx)*dy)/den;
  float t3 = 0.5f*logf(den/(4.f*sqrtf(d1*d2)+1e-8f));
  float bd = fminf(fmaxf(t1+t2+t3, 1e-8f), 100.f);
  float e  = fminf(fmaxf(1.f - expf(-bd), 0.f), 1.f);
  return 1.f - sqrtf(e);
}

__device__ __forceinline__ float bcef(float x, float t){
  return fmaxf(x,0.f) - x*t + log1pf(expf(-fabsf(x)));
}

// ---------------- K1c: extract GT boxes / classes / validity ----------------
__global__ __launch_bounds__(256) void k_gt(const float* __restrict__ labels,
                                            float* gtbox, float* validf, int* gtcls){
  int idx = blockIdx.x*256 + threadIdx.x;
  if (idx >= NB*NG) return;
  const float* l = labels + (size_t)idx*6;
  float s = l[0]+l[1]+l[2]+l[3]+l[4]+l[5];
  validf[idx] = (s > 0.f) ? 1.f : 0.f;
  gtcls[idx]  = (int)l[0];
  gtbox[idx*5+0]=l[1]; gtbox[idx*5+1]=l[2]; gtbox[idx*5+2]=l[3];
  gtbox[idx*5+3]=l[4]; gtbox[idx*5+4]=l[5];
}

// ---------------- K1: decode predictions, p-matrix, S = sum(-log1p(-p)) ----------------
__global__ __launch_bounds__(256) void k_decode(const float* __restrict__ o8,
                                                const float* __restrict__ o16,
                                                const float* __restrict__ o32,
                                                float* bbox, float* objL, float* Ssum, float* Pm){
  int idx = blockIdx.x*256 + threadIdx.x;           // exact NB*NA
  int b = idx / NA, a = idx % NA;
  int lvl,hw,w,ii; float st;
  anchor_geo(a,lvl,hw,w,ii,st);
  const float* src  = (lvl==0)?o8:((lvl==1)?o16:o32);
  const float* base = src + (size_t)b*21*hw + ii;   // channel c at base[c*hw]
  float gx = (float)(ii % w), gy = (float)(ii / w);
  float px = (base[0]            + gx)*st;
  float py = (base[(size_t)hw]   + gy)*st;
  float pw = expf(base[(size_t)2*hw])*st;
  float ph = expf(base[(size_t)3*hw])*st;
  float ang= base[(size_t)4*hw];
  float ob = base[(size_t)5*hw];
  float* bp = bbox + (size_t)idx*5;
  bp[0]=px; bp[1]=py; bp[2]=pw; bp[3]=ph; bp[4]=ang;
  objL[idx]=ob;
  float sob = 1.f/(1.f+expf(-ob));
  float S = 0.f;
  #pragma unroll
  for (int c=0;c<NCLS;c++){
    float z  = base[(size_t)(6+c)*hw];
    float sc = 1.f/(1.f+expf(-z));
    float p  = sqrtf(sc*sob);
    p = fminf(fmaxf(p, 1e-7f), 1.f - 1e-7f);
    Pm[((size_t)b*GP + c)*NA + a] = p;
    S += -log1pf(-p);
  }
  Pm[((size_t)b*GP + NCLS)*NA + a] = 0.f;           // pad class 15
  Ssum[idx] = S;
}

// ---------------- K1b: cand = any_g(in_box | in_ctr) ----------------
__global__ __launch_bounds__(256) void k_cand(const float* __restrict__ gtbox,
                                              const float* __restrict__ validf, float* candf){
  int idx = blockIdx.x*256 + threadIdx.x;
  int b = idx / NA, a = idx % NA;
  int lvl,hw,w,ii; float st;
  anchor_geo(a,lvl,hw,w,ii,st);
  float xc = ((float)(ii % w) + 0.5f)*st;
  float yc = ((float)(ii / w) + 0.5f)*st;
  float rad = 2.5f*st;
  bool cand=false;
  for (int g=0; g<NG; g++){
    const float* gt = gtbox + ((size_t)b*NG + g)*5;
    float v  = validf[b*NG+g];
    float dx = fabsf(xc - gt[0]), dy = fabsf(yc - gt[1]);
    bool ib = (dx < 0.5f*gt[2]) && (dy < 0.5f*gt[3]);
    bool ic = (dx < rad) && (dy < rad);
    cand = cand || ((ib||ic) && v>0.f);
  }
  candf[idx] = cand ? 1.f : 0.f;
}

// ---------------- K2: WMMA one-hot gather (pg) + cost + prob-IoU ----------------
// one wave = one 16(g) x 16(a) tile. pg tile = OneHot(16x16) x P^T(16x16) via WMMA,
// exact in f32 (1*p + 0*... under RNE). EXEC is all-ones through the WMMAs.
__global__ __launch_bounds__(256) void k_cost(const float* __restrict__ Pm,
                                              const float* __restrict__ bbox,
                                              const float* __restrict__ Ssum,
                                              const float* __restrict__ candf,
                                              const float* __restrict__ gtbox,
                                              const float* __restrict__ validf,
                                              const int*   __restrict__ gtcls,
                                              float* cost, float* iouA){
  int wid  = (blockIdx.x*256 + threadIdx.x) >> 5;   // global wave id, exact NB*4*1344
  int lane = threadIdx.x & 31;
  int at   = wid % A_TILES;
  int t2   = wid / A_TILES;
  int gt_t = t2 & 3;
  int b    = t2 >> 2;
  int g0 = gt_t*16, a0 = at*16;
  int halo = lane >> 4;           // 0: lanes 0-15, 1: lanes 16-31
  int l15  = lane & 15;
  int aCol = a0 + l15;
  int gRow = g0 + l15;
  int gc   = (gRow < NG) ? gtcls[b*NG + gRow] : -1;   // padded g rows -> zero one-hot
  const float* Pb = Pm + (size_t)b*GP*NA + aCol;

  v8f acc = {};
#if __has_builtin(__builtin_amdgcn_wmma_f32_16x16x4_f32)
  // A 16x4 f32: vgpr0 = K (lanes 0-15) / K+2 (lanes 16-31); vgpr1 = K+1 / K+3.
  // B 4x16 f32: mirrored row striping across lanes.
  #pragma unroll
  for (int k0=0; k0<16; k0+=4){
    int ka = k0 + halo*2;
    v2f Aop, Bop;
    Aop.x = (gc==ka)   ? 1.f : 0.f;
    Aop.y = (gc==ka+1) ? 1.f : 0.f;
    Bop.x = Pb[(size_t)ka    *NA];
    Bop.y = Pb[(size_t)(ka+1)*NA];
    acc = __builtin_amdgcn_wmma_f32_16x16x4_f32(false, Aop, false, Bop,
                                                (short)0, acc, false, false);
  }
#else
  // Fallback: confirmed f16 WMMA, K=32 with classes in K=0..15 and zero padding above.
  v16h Ah, Bh;
  #pragma unroll
  for (int j=0;j<8;j++){
    int kb = (j<4) ? (halo*8 + 2*j) : (16 + halo*8 + 2*(j-4));
    Ah[2*j]   = (_Float16)((gc==kb)   ? 1.f : 0.f);
    Ah[2*j+1] = (_Float16)((gc==kb+1) ? 1.f : 0.f);
    bool real = (kb < 16);
    Bh[2*j]   = real ? (_Float16)Pb[(size_t)kb    *NA] : (_Float16)0.f;
    Bh[2*j+1] = real ? (_Float16)Pb[(size_t)(kb+1)*NA] : (_Float16)0.f;
  }
  acc = __builtin_amdgcn_wmma_f32_16x16x32_f16(false, Ah, false, Bh,
                                               (short)0, acc, false, false);
#endif

  // per-lane anchor-side data (shared across the 8 g rows this lane holds)
  const float* bp = bbox + (size_t)(b*NA + aCol)*5;
  float bx=bp[0], by=bp[1], bw=bp[2], bh=bp[3], bt=bp[4];
  float Sv  = Ssum[b*NA + aCol];
  float cnd = candf[b*NA + aCol];
  int lvl,hw,w,ii; float st;
  anchor_geo(aCol,lvl,hw,w,ii,st);
  float xc = ((float)(ii%w)+0.5f)*st;
  float yc = ((float)(ii/w)+0.5f)*st;
  float rad = 2.5f*st;
  float pa,pb2,pc2; cov3(bw,bh,bt,pa,pb2,pc2);
  float pdet = fmaxf(pa*pb2 - pc2*pc2, 0.f);

  // C/D layout: lanes 0-15 -> M=v, lanes 16-31 -> M=v+8; N = lane&15
  #pragma unroll
  for (int v=0; v<8; v++){
    int g = g0 + v + halo*8;
    if (g >= NG) continue;
    const float* gt = gtbox + ((size_t)b*NG + g)*5;
    float vg = validf[b*NG + g];
    float ga,gb,gcv; cov3(gt[2],gt[3],gt[4],ga,gb,gcv);
    float gdet = fmaxf(ga*gb - gcv*gcv, 0.f);
    float iou = prob_iou_pre(gt[0],gt[1],ga,gb,gcv,gdet, bx,by,pa,pb2,pc2,pdet);
    iou = (vg>0.f) ? iou : 0.f;
    float pg   = acc[v];
    float clsc = Sv + log1pf(-pg) - logf(pg);
    float dx = fabsf(xc - gt[0]), dy = fabsf(yc - gt[1]);
    bool ib = (dx < 0.5f*gt[2]) && (dy < 0.5f*gt[3]) && (vg>0.f);
    bool ic = (dx < rad) && (dy < rad) && (vg>0.f);
    bool both = ib && ic;
    float cst = clsc + 3.f*(-logf(iou + 1e-8f))
              + (both      ? 0.f : 1e5f)
              + (cnd>0.f   ? 0.f : 1e6f)
              + (vg>0.f    ? 0.f : 1e9f);
    size_t off = ((size_t)(b*NG + g))*NA + aCol;
    cost[off] = cst;
    iouA[off] = iou;
  }
}

// ---------------- K3: per (b,g) dyn_k + k-th smallest cost threshold ----------------
// Emulates ranks=argsort(argsort(cost)) < dyn_k exactly: match iff
// cost < T, or cost == T and a <= L (stable tie cut by index).
__global__ __launch_bounds__(256) void k_select(const float* __restrict__ cost,
                                                const float* __restrict__ iouA,
                                                const float* __restrict__ candf,
                                                const float* __restrict__ validf,
                                                float* Tval, int* Lidx, int* dynk){
  const int bg = blockIdx.x;            // NB*NG blocks
  const int b  = bg / NG;
  const float* crow = cost + (size_t)bg*NA;
  const float* irow = iouA + (size_t)bg*NA;
  const float* cnd  = candf + (size_t)b*NA;
  const int tid = threadIdx.x;
  __shared__ float sm[256*10];
  __shared__ int   smi[256];
  __shared__ float sT;
  __shared__ int   sK, sMlt, sMeq, sPrev;

  // ---- dyn_k: sum of top-10 of iou*cand ----
  float loc[10];
  #pragma unroll
  for (int j=0;j<10;j++) loc[j] = -1.f;
  for (int a=tid; a<NA; a+=256){
    float v = irow[a]*cnd[a];
    if (v > loc[9]){
      int j=9;
      while (j>0 && v>loc[j-1]){ loc[j]=loc[j-1]; j--; }
      loc[j]=v;
    }
  }
  #pragma unroll
  for (int j=0;j<10;j++) sm[tid*10+j]=loc[j];
  __syncthreads();
  if (tid==0){
    float sum=0.f;
    for (int t=0;t<10;t++){
      float best=-2.f; int bi=0;
      for (int i=0;i<256*10;i++){ float v=sm[i]; if(v>best){best=v;bi=i;} }
      sum += best; sm[bi] = -2.f;
    }
    int k = (validf[bg] > 0.f) ? max((int)sum, 1) : 0;
    sK = min(k, 10);
  }
  __syncthreads();
  const int k = sK;
  if (k == 0){
    if (tid==0){ Tval[bg] = -3.4e38f; Lidx[bg] = -1; dynk[bg] = 0; }
    return;
  }

  // ---- k smallest costs -> T ----
  #pragma unroll
  for (int j=0;j<10;j++) loc[j] = 3.4e38f;
  for (int a=tid; a<NA; a+=256){
    float v = crow[a];
    if (v < loc[9]){
      int j=9;
      while (j>0 && v<loc[j-1]){ loc[j]=loc[j-1]; j--; }
      loc[j]=v;
    }
  }
  #pragma unroll
  for (int j=0;j<10;j++) sm[tid*10+j]=loc[j];
  __syncthreads();
  if (tid==0){
    float kth=0.f;
    for (int t=0;t<k;t++){
      float best=3.5e38f; int bi=0;
      for (int i=0;i<256*10;i++){ float v=sm[i]; if(v<best){best=v;bi=i;} }
      kth = best; sm[bi] = 3.5e38f;
    }
    sT = kth;
  }
  __syncthreads();
  const float T = sT;

  // ---- strict-less and tie counts ----
  int clt=0, ceq=0;
  for (int a=tid; a<NA; a+=256){
    float v = crow[a];
    clt += (v <  T);
    ceq += (v == T);
  }
  smi[tid]=clt; __syncthreads();
  if (tid==0){ int s=0; for(int i=0;i<256;i++) s+=smi[i]; sMlt=s; } __syncthreads();
  smi[tid]=ceq; __syncthreads();
  if (tid==0){ int s=0; for(int i=0;i<256;i++) s+=smi[i]; sMeq=s; } __syncthreads();
  const int r = k - sMlt;                  // ties to take, >= 1 by construction
  if (sMeq <= r){
    if (tid==0){ Tval[bg]=T; Lidx[bg]=NA; dynk[bg]=k; }
    return;
  }
  // find index of r-th tie (index order)
  int prev = -1;
  for (int it=0; it<r; it++){
    int lmin = 0x7fffffff;
    for (int a=tid; a<NA; a+=256)
      if (crow[a]==T && a>prev && a<lmin) lmin=a;
    smi[tid]=lmin; __syncthreads();
    if (tid==0){ int m=0x7fffffff; for(int i=0;i<256;i++) m=min(m,smi[i]); sPrev=m; }
    __syncthreads();
    prev = sPrev;
  }
  if (tid==0){ Tval[bg]=T; Lidx[bg]=prev; dynk[bg]=k; }
}

// ---------------- K4: per-anchor matching, conflict resolution, losses ----------------
__global__ __launch_bounds__(256) void k_assign(const float* __restrict__ cost,
                                                const float* __restrict__ iouA,
                                                const float* __restrict__ bbox,
                                                const float* __restrict__ objL,
                                                const float* __restrict__ gtbox,
                                                const int*   __restrict__ gtcls,
                                                const float* __restrict__ Tval,
                                                const int*   __restrict__ Lidx,
                                                const int*   __restrict__ dynk,
                                                const float* __restrict__ o8,
                                                const float* __restrict__ o16,
                                                const float* __restrict__ o32,
                                                float* part){
  __shared__ float sT[NG]; __shared__ int sL[NG]; __shared__ int sKk[NG];
  __shared__ float red[256];
  int idx = blockIdx.x*256 + threadIdx.x;           // exact NB*NA; 84 blocks per b
  int b = idx / NA, a = idx % NA;
  if (threadIdx.x < NG){
    sT[threadIdx.x]  = Tval[b*NG + threadIdx.x];
    sL[threadIdx.x]  = Lidx[b*NG + threadIdx.x];
    sKk[threadIdx.x] = dynk[b*NG + threadIdx.x];
  }
  __syncthreads();

  int cnt=0, firstg=-1, ming=0; float cmin=3.4e38f;
  for (int g=0; g<NG; g++){
    float c = cost[((size_t)(b*NG + g))*NA + a];
    bool m = (sKk[g]>0) && ((c < sT[g]) || (c == sT[g] && a <= sL[g]));
    if (m){ cnt++; if (firstg<0) firstg=g; }
    if (c < cmin){ cmin=c; ming=g; }                 // first-occurrence argmin
  }
  int   gsel = (cnt>1) ? ming : firstg;              // conflict -> argmin_g cost
  float fgf  = (cnt>0) ? 1.f : 0.f;
  float liou = 0.f, lcls = 0.f;
  float ob   = objL[idx];
  if (cnt > 0){
    float piou = iouA[((size_t)(b*NG + gsel))*NA + a];
    const float* gt = gtbox + ((size_t)b*NG + gsel)*5;
    const float* bp = bbox + (size_t)idx*5;
    float pa,pb2,pc2; cov3(bp[2],bp[3],bp[4],pa,pb2,pc2);
    float pdet = fmaxf(pa*pb2-pc2*pc2, 0.f);
    float ga,gb,gcv; cov3(gt[2],gt[3],gt[4],ga,gb,gcv);
    float gdet = fmaxf(ga*gb-gcv*gcv, 0.f);
    liou = 1.f - prob_iou_pre(bp[0],bp[1],pa,pb2,pc2,pdet, gt[0],gt[1],ga,gb,gcv,gdet);
    int mc = gtcls[b*NG + gsel];
    int lvl,hw,w,ii; float st;
    anchor_geo(a,lvl,hw,w,ii,st);
    const float* src  = (lvl==0)?o8:((lvl==1)?o16:o32);
    const float* base = src + (size_t)b*21*hw + ii;
    #pragma unroll
    for (int c=0;c<NCLS;c++){
      float z   = base[(size_t)(6+c)*hw];
      float tgt = (c==mc) ? piou : 0.f;
      lcls += bcef(z, tgt);
    }
  }
  float lobj = bcef(ob, fgf);

  // deterministic block reduction -> per-block partials (no float atomics)
  float vals[4] = { fgf, liou, lobj, lcls };
  #pragma unroll
  for (int q=0;q<4;q++){
    red[threadIdx.x] = vals[q]; __syncthreads();
    for (int s2=128; s2>0; s2>>=1){
      if (threadIdx.x < s2) red[threadIdx.x] += red[threadIdx.x+s2];
      __syncthreads();
    }
    if (threadIdx.x==0) part[(size_t)blockIdx.x*4 + q] = red[0];
    __syncthreads();
  }
}

// ---------------- K5: finalize 6 outputs ----------------
__global__ void k_final(const float* __restrict__ part, const float* __restrict__ validf,
                        float* out){
  float nfg=0.f, si=0.f, so=0.f, sc=0.f;
  for (int i=0;i<NBLK_BA;i++){
    nfg += part[i*4+0]; si += part[i*4+1]; so += part[i*4+2]; sc += part[i*4+3];
  }
  float ng=0.f;
  for (int i=0;i<NB*NG;i++) ng += validf[i];
  float nf = fmaxf(nfg, 1.f);
  float li = si/nf, lo = so/nf, lc = sc/nf;
  out[0] = 5.f*li + lo + lc;
  out[1] = 5.f*li;
  out[2] = lo;
  out[3] = lc;
  out[4] = 0.f;
  out[5] = nfg / fmaxf(ng, 1.f);
}

// ---------------- host launcher ----------------
extern "C" void kernel_launch(void* const* d_in, const int* in_sizes, int n_in,
                              void* d_out, int out_size, void* d_ws, size_t ws_size,
                              hipStream_t stream){
  const float* o8     = (const float*)d_in[0];
  const float* o16    = (const float*)d_in[1];
  const float* o32    = (const float*)d_in[2];
  const float* labels = (const float*)d_in[3];
  float* out = (float*)d_out;

  const size_t BA  = (size_t)NB*NA;        // 172032
  const size_t BGA = (size_t)NB*NG*NA;     // 10321920
  // workspace carve-up (all 4-byte elements)
  float* ws    = (float*)d_ws;
  float* bbox  = ws;                               // BA*5
  float* objL  = bbox  + BA*5;                     // BA
  float* Ssum  = objL  + BA;                       // BA
  float* Pm    = Ssum  + BA;                       // NB*GP*NA
  float* candf = Pm    + (size_t)NB*GP*NA;         // BA
  float* gtbox = candf + BA;                       // NB*NG*5
  float* validf= gtbox + (size_t)NB*NG*5;          // NB*NG
  int*   gtcls = (int*)(validf + NB*NG);           // NB*NG
  float* Tval  = (float*)(gtcls + NB*NG);          // NB*NG
  int*   Lidx  = (int*)(Tval + NB*NG);             // NB*NG
  int*   dynk  = Lidx + NB*NG;                     // NB*NG
  float* part  = (float*)(dynk + NB*NG);           // NBLK_BA*4
  float* cost  = part + (size_t)NBLK_BA*4;         // BGA
  float* iouA  = cost + BGA;                       // BGA
  size_t need_bytes = ((size_t)(iouA + BGA) - (size_t)ws);
  if (ws_size < need_bytes) return;                // not enough scratch: bail safely

  k_gt    <<<(NB*NG + 255)/256, 256, 0, stream>>>(labels, gtbox, validf, gtcls);
  k_decode<<<NBLK_BA, 256, 0, stream>>>(o8, o16, o32, bbox, objL, Ssum, Pm);
  k_cand  <<<NBLK_BA, 256, 0, stream>>>(gtbox, validf, candf);
  // NB * 4 g-tiles * 1344 a-tiles waves, 8 waves per block
  k_cost  <<<(NB*4*A_TILES)/8, 256, 0, stream>>>(Pm, bbox, Ssum, candf, gtbox, validf, gtcls,
                                                 cost, iouA);
  k_select<<<NB*NG, 256, 0, stream>>>(cost, iouA, candf, validf, Tval, Lidx, dynk);
  k_assign<<<NBLK_BA, 256, 0, stream>>>(cost, iouA, bbox, objL, gtbox, gtcls,
                                        Tval, Lidx, dynk, o8, o16, o32, part);
  k_final <<<1, 1, 0, stream>>>(part, validf, out);
}
